// Step3Attention_21423296872625
// MI455X (gfx1250) — matrix operand order repaired and checked by reading, not compiled, not run on hardware
//
#include <hip/hip_runtime.h>
#include <stdint.h>

// ---------------------------------------------------------------------------
// Step3Attention for MI455X (gfx1250, wave32, WMMA).
// [convert (+weight transpose)] -> [QKV GEMM bf16 WMMA, async double-buffer]
// -> [RoPE+split, V pre-transposed] -> [flash attention bf16 WMMA, async
//    double-buffered K/V pipeline] -> [out GEMM bf16 WMMA + bias + nan_to_num].
// All tile staging: contiguous 16B GLOBAL_LOAD_ASYNC_TO_LDS_B128 (ASYNCcnt).
// Dynamic LDS (WGP pool is 320KB) to allow BK=64 double buffering.
// ---------------------------------------------------------------------------

#define B_    2
#define T_    2048
#define D_    2048
#define H_    32
#define KVH_  8
#define HD_   64
#define NQKV_ 3072          // H*HD + 2*KVH*HD
#define MTOK_ 4096          // B*T

typedef __attribute__((ext_vector_type(16))) __bf16 v16bf;
typedef __attribute__((ext_vector_type(8)))  float  v8f;

__device__ __forceinline__ uint16_t f2bf(float x) {
  uint32_t u = __float_as_uint(x);
  u += 0x7FFFu + ((u >> 16) & 1u);      // round-to-nearest-even
  return (uint16_t)(u >> 16);
}

__device__ __forceinline__ v8f vzero() {
  v8f z;
#pragma unroll
  for (int i = 0; i < 8; ++i) z[i] = 0.0f;
  return z;
}

// 16B async tile copy global->LDS (CDNA5 direct-to-LDS DMA, ASYNCcnt-tracked).
__device__ __forceinline__ void cp16(void* lds, const void* g) {
  uint32_t l = (uint32_t)(uintptr_t)lds;   // generic LDS ptr low 32 = offset
  uint64_t a = (uint64_t)(uintptr_t)g;
  asm volatile("global_load_async_to_lds_b128 %0, %1, off"
               :: "v"(l), "v"(a) : "memory");
}
__device__ __forceinline__ void cp_wait() {
  asm volatile("s_wait_asynccnt 0x0" ::: "memory");
}

// Build a 16-element bf16 fragment from two 16B LDS reads (ds_load_b128).
__device__ __forceinline__ v16bf ld_frag(const uint16_t* p0, const uint16_t* p1) {
  uint4 t0 = *(const uint4*)p0;
  uint4 t1 = *(const uint4*)p1;
  v16bf r;
  __builtin_memcpy(&r, &t0, 16);
  __builtin_memcpy((char*)&r + 16, &t1, 16);
  return r;
}

// A-fragment (16x32 MxK): lane row = lane&15; element i -> K=(i&7)+16*(i>>3)+8*half.
__device__ __forceinline__ v16bf ld_a(const uint16_t* base, int row, int stride,
                                      int kk, int half) {
  const uint16_t* p = base + row * stride + kk + 8 * half;
  return ld_frag(p, p + 16);
}
// B-fragment (32x16 KxN): lane col = lane&15; element i -> K=(i&7)+16*(i>>3)+8*half.
// base is [col][K] (K contiguous).
__device__ __forceinline__ v16bf ld_b(const uint16_t* base, int col, int stride,
                                      int kk, int half) {
  const uint16_t* p = base + col * stride + kk + 8 * half;
  return ld_frag(p, p + 16);
}

// ---------------------------------------------------------------------------
// Kernel 0a: plain f32 -> bf16 convert (n multiple of 4).
// ---------------------------------------------------------------------------
__global__ void __launch_bounds__(256) k_f32_to_bf16(const float* __restrict__ in,
                                                     uint16_t* __restrict__ out,
                                                     int n) {
  int i = (blockIdx.x * 256 + threadIdx.x) * 4;
  if (i >= n) return;
  float4 v = *(const float4*)(in + i);
  uint2 o;
  o.x = (uint32_t)f2bf(v.x) | ((uint32_t)f2bf(v.y) << 16);
  o.y = (uint32_t)f2bf(v.z) | ((uint32_t)f2bf(v.w) << 16);
  *(uint2*)(out + i) = o;
}

// ---------------------------------------------------------------------------
// Kernel 0b: f32 [K][N] -> bf16 [N][K] transpose-convert via 64x64 LDS tile.
// ---------------------------------------------------------------------------
__global__ void __launch_bounds__(256) k_convert_T(const float* __restrict__ in,
                                                   uint16_t* __restrict__ out,
                                                   int K, int N) {
  __shared__ __align__(16) uint16_t Ts[64 * 72];
  const int tid = threadIdx.x;
  const int n0 = blockIdx.x * 64;
  const int k0 = blockIdx.y * 64;
#pragma unroll
  for (int i = 0; i < 4; ++i) {
    int c = tid + 256 * i;              // 1024 chunks of 4 f32
    int k = c >> 4, nc = (c & 15) * 4;
    float4 v = *(const float4*)&in[(size_t)(k0 + k) * N + n0 + nc];
    Ts[(nc + 0) * 72 + k] = f2bf(v.x);
    Ts[(nc + 1) * 72 + k] = f2bf(v.y);
    Ts[(nc + 2) * 72 + k] = f2bf(v.z);
    Ts[(nc + 3) * 72 + k] = f2bf(v.w);
  }
  __syncthreads();
#pragma unroll
  for (int i = 0; i < 2; ++i) {
    int c = tid + 256 * i;              // 512 chunks of 8 bf16
    int n = c >> 3, kc = (c & 7) * 8;
    *(uint4*)&out[(size_t)(n0 + n) * K + k0 + kc] = *(const uint4*)&Ts[n * 72 + kc];
  }
}

// ---------------------------------------------------------------------------
// Kernel 1/4: C[M,N] = A[M,K](bf16) @ Bt[N,K](bf16)^T (+bias, +nan_to_num).
// 128x128x64 tiles, 8 waves (64x32 per wave), async double-buffered dynamic
// LDS (2*(128*72)*2 matrices = 72KB). 16 WMMAs per stage, 32 stages.
// ---------------------------------------------------------------------------
#define BM 128
#define BN 128
#define BK 64
#define TSTR 72                   // padded LDS stride
#define GTILE (BM * TSTR)         // elems per buffer per matrix
#define GEMM_LDS_BYTES (4 * GTILE * 2)   // As[2] + Bs[2], bf16

template <bool BIAS, bool CLAMP>
__global__ void __launch_bounds__(256) k_gemm_bf16(
    const uint16_t* __restrict__ A, const uint16_t* __restrict__ Bt,
    const float* __restrict__ bias, float* __restrict__ C,
    int M, int N, int K) {
  extern __shared__ __align__(16) uint16_t smem_g[];
  uint16_t* As = smem_g;                 // [2][GTILE]
  uint16_t* Bs = smem_g + 2 * GTILE;     // [2][GTILE]

  const int tid  = threadIdx.x;
  const int lane = tid & 31;
  const int wave = tid >> 5;
  const int half = lane >> 4;
  const int l16  = lane & 15;
  const int m0   = blockIdx.y * BM;
  const int n0   = blockIdx.x * BN;
  const int wm   = wave >> 2;   // 0..1 -> 64-row slab
  const int wn   = wave & 3;    // 0..3 -> 32-col slab

  v8f acc[4][2];
#pragma unroll
  for (int a = 0; a < 4; ++a)
#pragma unroll
    for (int b = 0; b < 2; ++b) acc[a][b] = vzero();

  const int nstage = K / BK;

  // stage(buf, k0): 128 rows x 64 K for both matrices, 4+4 async 16B chunks.
#define STAGE(bufi, kk0)                                                       \
  do {                                                                         \
    _Pragma("unroll")                                                          \
    for (int i = 0; i < 4; ++i) {                                              \
      int c = tid + 256 * i, row = c >> 3, cc = (c & 7) * 8;                   \
      cp16(&As[(bufi) * GTILE + row * TSTR + cc],                              \
           &A[(size_t)(m0 + row) * K + (kk0) + cc]);                           \
      cp16(&Bs[(bufi) * GTILE + row * TSTR + cc],                              \
           &Bt[(size_t)(n0 + row) * K + (kk0) + cc]);                          \
    }                                                                          \
  } while (0)

  STAGE(0, 0);

  for (int s = 0; s < nstage; ++s) {
    const int buf = s & 1;
    cp_wait();
    __syncthreads();
    if (s + 1 < nstage) STAGE(buf ^ 1, (s + 1) * BK);

    const uint16_t* Ab = &As[buf * GTILE];
    const uint16_t* Bb = &Bs[buf * GTILE];
#pragma unroll
    for (int kk = 0; kk < BK; kk += 32) {
      v16bf bf[2];
#pragma unroll
      for (int tn = 0; tn < 2; ++tn)
        bf[tn] = ld_b(Bb, wn * 32 + tn * 16 + l16, TSTR, kk, half);
#pragma unroll
      for (int tm = 0; tm < 4; ++tm) {
        v16bf af = ld_a(Ab, wm * 64 + tm * 16 + l16, TSTR, kk, half);
#pragma unroll
        for (int tn = 0; tn < 2; ++tn)
          acc[tm][tn] = __builtin_amdgcn_wmma_f32_16x16x32_bf16(
              false, af, false, bf[tn], (short)0, acc[tm][tn], false, false);
      }
    }
  }
#undef STAGE

  // Epilogue: lane col = lane&15, rows = vgpr + 8*half.
#pragma unroll
  for (int tm = 0; tm < 4; ++tm) {
#pragma unroll
    for (int tn = 0; tn < 2; ++tn) {
      int col  = n0 + wn * 32 + tn * 16 + l16;
      float bv = BIAS ? bias[col] : 0.0f;
#pragma unroll
      for (int r = 0; r < 8; ++r) {
        int row = m0 + wm * 64 + tm * 16 + 8 * half + r;
        float v = acc[tm][tn][r] + bv;
        if (CLAMP) {
          if (!(v == v)) v = 0.0f;
          else if (v > 1.0e38f) v = 10000.0f;
          else if (v < -1.0e38f) v = -10000.0f;
        }
        C[(size_t)row * N + col] = v;
      }
    }
  }
}

// ---------------------------------------------------------------------------
// Kernel 2: RoPE + head split. One block per (b,t) token.
// Q carries the 1/sqrt(HD)=0.125 softmax scale (exact in bf16).
// Qb[B][H][T][HD], Kb[B][KVH][T][HD], VtG[B][KVH][HD][T] (V pre-transposed).
// ---------------------------------------------------------------------------
__global__ void __launch_bounds__(256) k_rope_split(
    const float* __restrict__ qkv, const float* __restrict__ rcos,
    const float* __restrict__ rsin, uint16_t* __restrict__ Qb,
    uint16_t* __restrict__ Kb, uint16_t* __restrict__ VtG) {
  const int idx = blockIdx.x;              // b*T + t
  const int b   = idx / T_;
  const int t   = idx - b * T_;
  const int tid = threadIdx.x;
  const float* row = qkv + (size_t)idx * NQKV_;
  const float* cr  = rcos + (size_t)t * (HD_ / 2);
  const float* sr  = rsin + (size_t)t * (HD_ / 2);

  for (int p = tid; p < H_ * (HD_ / 2); p += 256) {   // Q rotary pairs
    int h = p >> 5, pp = p & 31;
    float xe = row[h * HD_ + 2 * pp];
    float xo = row[h * HD_ + 2 * pp + 1];
    float c = cr[pp], s = sr[pp];
    size_t o = ((size_t)(b * H_ + h) * T_ + t) * HD_ + 2 * pp;
    Qb[o]     = f2bf((xe * c - xo * s) * 0.125f);
    Qb[o + 1] = f2bf((xe * s + xo * c) * 0.125f);
  }
  for (int p = tid; p < KVH_ * (HD_ / 2); p += 256) { // K rotary pairs
    int h = p >> 5, pp = p & 31;
    float xe = row[H_ * HD_ + h * HD_ + 2 * pp];
    float xo = row[H_ * HD_ + h * HD_ + 2 * pp + 1];
    float c = cr[pp], s = sr[pp];
    size_t o = ((size_t)(b * KVH_ + h) * T_ + t) * HD_ + 2 * pp;
    Kb[o]     = f2bf(xe * c - xo * s);
    Kb[o + 1] = f2bf(xe * s + xo * c);
  }
  for (int e = tid; e < KVH_ * HD_; e += 256) {       // V, transposed store
    int h = e >> 6, d = e & 63;
    float v = row[H_ * HD_ + KVH_ * HD_ + h * HD_ + d];
    VtG[((size_t)(b * KVH_ + h) * HD_ + d) * T_ + t] = f2bf(v);
  }
}

// ---------------------------------------------------------------------------
// Kernel 3: causal flash attention, one block per (b, h, 64-row q-block).
// Wave w: q-tile qm=w&3; S phase kn in {2*(w>>2),+1}; O phase dn likewise.
// K/V tiles double-buffered: next kv-block's async DMA overlaps S-WMMA,
// softmax and O-WMMA of the current one. 3 barriers per kv iteration.
// ---------------------------------------------------------------------------
#define KSTR 72                 // padded bf16 LDS stride
#define SSTR 68                 // padded f32 LDS stride
#define ATILE (64 * KSTR)       // elems per bf16 tile
// LDS carve: Qs | Ks[2] | Vt[2] | Ps | Ss(f32) | alphas | linv
#define ATT_OFF_QS   0
#define ATT_OFF_KS   (ATILE)
#define ATT_OFF_VT   (3 * ATILE)
#define ATT_OFF_PS   (5 * ATILE)
#define ATT_BYTES_BF (6 * ATILE * 2)
#define ATT_LDS_BYTES (ATT_BYTES_BF + 64 * SSTR * 4 + 2 * 64 * 4)

__global__ void __launch_bounds__(256) k_flash_attn(
    const uint16_t* __restrict__ Qb, const uint16_t* __restrict__ Kb,
    const uint16_t* __restrict__ VtG, uint16_t* __restrict__ AO) {
  extern __shared__ __align__(16) uint16_t smem_a[];
  uint16_t* Qs = smem_a + ATT_OFF_QS;
  uint16_t* Ks = smem_a + ATT_OFF_KS;    // [2][ATILE]
  uint16_t* Vt = smem_a + ATT_OFF_VT;    // [2][ATILE], layout [hd][kv]
  uint16_t* Ps = smem_a + ATT_OFF_PS;    // P bf16 [q][kv]
  float*    Ss     = (float*)(smem_a + 6 * ATILE);   // [64][SSTR]
  float*    alphas = Ss + 64 * SSTR;
  float*    linv   = alphas + 64;

  const int q0  = blockIdx.x * 64;
  const int h   = blockIdx.y;
  const int b   = blockIdx.z;
  const int kvh = h >> 2;                  // G = H/KVH = 4
  const int tid  = threadIdx.x;
  const int lane = tid & 31;
  const int wave = tid >> 5;
  const int half = lane >> 4;
  const int l16  = lane & 15;
  const int qm   = wave & 3;
  const int wp   = wave >> 2;              // 0..1

  const uint16_t* Qg = Qb  + ((size_t)(b * H_ + h) * T_) * HD_;
  const uint16_t* Kg = Kb  + ((size_t)(b * KVH_ + kvh) * T_) * HD_;
  const uint16_t* Vg = VtG + ((size_t)(b * KVH_ + kvh) * HD_) * T_;

  // Stage Q (64x64), contiguous async 16B chunks.
#pragma unroll
  for (int i = 0; i < 2; ++i) {
    int c = tid + 256 * i, row = c >> 3, cc = (c & 7) * 8;
    cp16(&Qs[row * KSTR + cc], &Qg[(size_t)(q0 + row) * HD_ + cc]);
  }
  cp_wait();
  __syncthreads();

  v16bf qa[2];
#pragma unroll
  for (int hk = 0; hk < 2; ++hk)
    qa[hk] = ld_a(Qs, qm * 16 + l16, KSTR, hk * 32, half);

  float m_run = -1.0e30f, l_run = 0.0f;
  v8f oacc[2] = {vzero(), vzero()};

#define STAGE_KV(bufi, kb)                                                     \
  do {                                                                         \
    _Pragma("unroll")                                                          \
    for (int i = 0; i < 2; ++i) {                                              \
      int c = tid + 256 * i, row = c >> 3, cc = (c & 7) * 8;                   \
      cp16(&Ks[(bufi) * ATILE + row * KSTR + cc],                              \
           &Kg[(size_t)((kb) + row) * HD_ + cc]);                              \
      cp16(&Vt[(bufi) * ATILE + row * KSTR + cc],                              \
           &Vg[(size_t)row * T_ + (kb) + cc]);                                 \
    }                                                                          \
  } while (0)

  const int nit = q0 / 64 + 1;
  STAGE_KV(0, 0);

  for (int it = 0; it < nit; ++it) {
    const int kb  = it * 64;
    const int buf = it & 1;
    cp_wait();
    __syncthreads();
    if (it + 1 < nit) STAGE_KV(buf ^ 1, kb + 64);

    const uint16_t* Kt = &Ks[buf * ATILE];
    const uint16_t* Vb = &Vt[buf * ATILE];

    // S = Q K^T (pre-scaled), causal mask, -> Ss.
#pragma unroll
    for (int t = 0; t < 2; ++t) {
      int kn = 2 * wp + t;
      v8f s = vzero();
#pragma unroll
      for (int hk = 0; hk < 2; ++hk) {
        v16bf kf = ld_b(Kt, kn * 16 + l16, KSTR, hk * 32, half);
        s = __builtin_amdgcn_wmma_f32_16x16x32_bf16(false, qa[hk], false, kf,
                                                    (short)0, s, false, false);
      }
      int col = kn * 16 + l16;
#pragma unroll
      for (int r = 0; r < 8; ++r) {
        int row = qm * 16 + 8 * half + r;
        float v = s[r];
        if (kb + col > q0 + row) v = -1.0e30f;
        Ss[row * SSTR + col] = v;
      }
    }
    __syncthreads();

    // Online softmax: thread r<64 owns q-row r (vectorized f32x4 LDS reads).
    if (tid < 64) {
      const float* srow = &Ss[tid * SSTR];
      float mx = m_run;
#pragma unroll
      for (int j = 0; j < 64; j += 4) {
        float4 s4 = *(const float4*)&srow[j];
        mx = fmaxf(mx, fmaxf(fmaxf(s4.x, s4.y), fmaxf(s4.z, s4.w)));
      }
      float alpha = __expf(m_run - mx);
      float sum = 0.0f;
      uint16_t* prow = &Ps[tid * KSTR];
#pragma unroll
      for (int j = 0; j < 64; j += 4) {
        float4 s4 = *(const float4*)&srow[j];
        float p0 = __expf(s4.x - mx), p1 = __expf(s4.y - mx);
        float p2 = __expf(s4.z - mx), p3 = __expf(s4.w - mx);
        sum += (p0 + p1) + (p2 + p3);
        uint2 pk;
        pk.x = (uint32_t)f2bf(p0) | ((uint32_t)f2bf(p1) << 16);
        pk.y = (uint32_t)f2bf(p2) | ((uint32_t)f2bf(p3) << 16);
        *(uint2*)&prow[j] = pk;
      }
      l_run = l_run * alpha + sum;
      m_run = mx;
      alphas[tid] = alpha;
    }
    __syncthreads();

    // O = O*alpha + P V.
    float ar[8];
#pragma unroll
    for (int r = 0; r < 8; ++r) ar[r] = alphas[qm * 16 + 8 * half + r];
#pragma unroll
    for (int t = 0; t < 2; ++t) {
      int dn = 2 * wp + t;
#pragma unroll
      for (int r = 0; r < 8; ++r) oacc[t][r] *= ar[r];
#pragma unroll
      for (int kk = 0; kk < 64; kk += 32) {
        v16bf pa = ld_a(Ps, qm * 16 + l16, KSTR, kk, half);
        v16bf vf = ld_b(Vb, dn * 16 + l16, KSTR, kk, half);
        oacc[t] = __builtin_amdgcn_wmma_f32_16x16x32_bf16(
            false, pa, false, vf, (short)0, oacc[t], false, false);
      }
    }
    // No end barrier: top-of-iteration barrier orders Ss/Ps reuse.
  }
#undef STAGE_KV

  __syncthreads();
  if (tid < 64) linv[tid] = (l_run > 0.0f) ? (1.0f / l_run) : 0.0f;
  __syncthreads();

#pragma unroll
  for (int t = 0; t < 2; ++t) {
    int d = (2 * wp + t) * 16 + l16;
#pragma unroll
    for (int r = 0; r < 8; ++r) {
      int row = qm * 16 + 8 * half + r;
      float v = oacc[t][r] * linv[row];
      AO[((size_t)(b * T_ + q0 + row)) * D_ + h * HD_ + d] = f2bf(v);
    }
  }
}

// ---------------------------------------------------------------------------
// Host launch. Workspace layout (MB offsets):
//   [0,16)  xq bf16        [16,28) W_qkv^T bf16   [28,36)  W_out^T bf16
//   [36,84) qkv f32        [84,100) Q bf16        [100,104) K bf16
//   [104,108) V^T bf16     [108,124) AO bf16
// ---------------------------------------------------------------------------
extern "C" void kernel_launch(void* const* d_in, const int* in_sizes, int n_in,
                              void* d_out, int out_size, void* d_ws, size_t ws_size,
                              hipStream_t stream) {
  (void)in_sizes; (void)n_in; (void)out_size; (void)ws_size;
  const float* x_q   = (const float*)d_in[0];
  const float* W_qkv = (const float*)d_in[1];
  const float* W_out = (const float*)d_in[2];
  const float* b_out = (const float*)d_in[3];
  const float* rcos  = (const float*)d_in[4];
  const float* rsin  = (const float*)d_in[5];

  char* ws = (char*)d_ws;
  const size_t MB = 1024ull * 1024ull;
  uint16_t* xq_bf    = (uint16_t*)(ws + 0 * MB);
  uint16_t* wqkvT_bf = (uint16_t*)(ws + 16 * MB);
  uint16_t* woutT_bf = (uint16_t*)(ws + 28 * MB);
  float*    qkv_f    = (float*)   (ws + 36 * MB);
  uint16_t* Qb       = (uint16_t*)(ws + 84 * MB);
  uint16_t* Kb       = (uint16_t*)(ws + 100 * MB);
  uint16_t* Vt       = (uint16_t*)(ws + 104 * MB);
  uint16_t* AO       = (uint16_t*)(ws + 108 * MB);

  const int n_xq = MTOK_ * D_;    // 8388608

  k_f32_to_bf16<<<n_xq / 1024, 256, 0, stream>>>(x_q, xq_bf, n_xq);
  k_convert_T<<<dim3(NQKV_ / 64, D_ / 64), 256, 0, stream>>>(W_qkv, wqkvT_bf,
                                                             D_, NQKV_);
  k_convert_T<<<dim3(D_ / 64, D_ / 64), 256, 0, stream>>>(W_out, woutT_bf,
                                                          D_, D_);

  k_gemm_bf16<false, false>
      <<<dim3(NQKV_ / BN, MTOK_ / BM), 256, GEMM_LDS_BYTES, stream>>>(
          xq_bf, wqkvT_bf, nullptr, qkv_f, MTOK_, NQKV_, D_);

  k_rope_split<<<MTOK_, 256, 0, stream>>>(qkv_f, rcos, rsin, Qb, Kb, Vt);

  k_flash_attn<<<dim3(T_ / 64, H_, B_), 256, ATT_LDS_BYTES, stream>>>(
      Qb, Kb, Vt, AO);

  k_gemm_bf16<true, true>
      <<<dim3(D_ / BN, MTOK_ / BM), 256, GEMM_LDS_BYTES, stream>>>(
          AO, woutT_bf, b_out, (float*)d_out, MTOK_, D_, D_);
}
